// GRU_66219805770266
// MI455X (gfx1250) — matrix operand order repaired and checked by reading, not compiled, not run on hardware
//
#include <hip/hip_runtime.h>
#include <hip/hip_bf16.h>
#include <math.h>

// GRU: L=2 independent layers, D=H=1024, S=512, B=32.
// Phase 1: big bf16-WMMA GEMMs for input projections (parallel, compute bound).
// Phase 2: persistent kernel, 512 sequential steps, grid barrier via atomics,
//          recurrent U-GEMMs on v_wmma_f32_16x16x32_bf16, state kept in f32.

#define L_  2
#define D_  1024
#define H_  1024
#define S_  512
#define B_  32
#define SB_ (S_ * B_)

typedef __attribute__((ext_vector_type(8)))  __bf16 v8bf;
typedef __attribute__((ext_vector_type(16))) __bf16 v16bf;
typedef __attribute__((ext_vector_type(8)))  float  v8f;

#define WMMA_BF16(a, b, c) \
  __builtin_amdgcn_wmma_f32_16x16x32_bf16(false, (a), false, (b), (short)0, (c), false, false)

// A-matrix fragment (16x32, 16-bit): lane m = lane%16; half=lane/16.
// element e<8 : K = kc + e + 8*half ; e>=8 : K = kc + (e+8) + 8*half.
__device__ __forceinline__ v16bf load_fragA(const __bf16* base, int ld, int row, int kc, int half) {
  const __bf16* p = base + (size_t)row * ld + kc + half * 8;
  v8bf lo = *(const v8bf*)(p);        // K = kc+8h .. +7
  v8bf hi = *(const v8bf*)(p + 16);   // K = kc+16+8h .. +7
  return __builtin_shufflevector(lo, hi, 0,1,2,3,4,5,6,7,8,9,10,11,12,13,14,15);
}

// B-matrix fragment (32x16, 16-bit): lane n = lane%16; e<8: K = kc + e + 8*half; e>=8: K = kc + (e+8) + 8*half.
// B(k,n) is read from row-major U[n][k], i.e. out = A * U^T.
__device__ __forceinline__ v16bf load_fragB(const __bf16* base, int ld, int row, int kc, int half) {
  const __bf16* p = base + (size_t)row * ld + kc + half * 8;
  v8bf lo = *(const v8bf*)(p);
  v8bf hi = *(const v8bf*)(p + 16);
  return __builtin_shufflevector(lo, hi, 0,1,2,3,4,5,6,7,8,9,10,11,12,13,14,15);
}

// ---------------- prep kernels ----------------
__global__ void cvt_bf16_kernel(const float* __restrict__ in, __bf16* __restrict__ out, int n) {
  int i = blockIdx.x * blockDim.x + threadIdx.x;
  int stride = gridDim.x * blockDim.x;
  for (; i < n; i += stride) out[i] = (__bf16)in[i];
}

__global__ void init_h_kernel(const float* __restrict__ h0, float* __restrict__ hf,
                              __bf16* __restrict__ hb, int n) {
  int i = blockIdx.x * blockDim.x + threadIdx.x;
  int stride = gridDim.x * blockDim.x;
  for (; i < n; i += stride) { float v = h0[i]; hf[i] = v; hb[i] = (__bf16)v; }
}

// ---------------- input projection GEMM ----------------
// xproj[g][l][s*B+b][h] = bf16( sum_d x[s,b,d]*W_g[l][h][d] + b_g[l][h] )
// per wave: 32x32 output tile, K loop in steps of 32 (4 wmma / iter).
__global__ __launch_bounds__(256) void proj_kernel(
    const __bf16* __restrict__ xbf,     // [SB][D]
    const __bf16* __restrict__ Wbf,     // [3][L][H][D]
    const float* __restrict__ bz, const float* __restrict__ br, const float* __restrict__ bh,
    __bf16* __restrict__ xproj)         // [3][L][SB][H]
{
  const int tid = threadIdx.x, wave = tid >> 5, lane = tid & 31;
  const int half = lane >> 4, lr = lane & 15;
  int gw = blockIdx.x * 8 + wave;          // 0 .. 98303
  int nt = gw & 31;        gw >>= 5;       // 32 N-tiles of 32
  int mt = gw & 511;       gw >>= 9;       // 512 M-tiles of 32
  int l  = gw & 1;         gw >>= 1;       // layer
  int g  = gw;                             // gate 0..2

  const __bf16* A  = xbf + (size_t)mt * 32 * D_;
  const __bf16* B0 = Wbf + (size_t)(g * L_ + l) * H_ * D_ + (size_t)(nt * 32) * D_;

  v8f acc[2][2] = {};
  for (int kc = 0; kc < D_; kc += 32) {
    v16bf a0 = load_fragA(A,            D_, lr, kc, half);
    v16bf a1 = load_fragA(A + 16 * D_,  D_, lr, kc, half);
    v16bf b0 = load_fragB(B0,           D_, lr, kc, half);
    v16bf b1 = load_fragB(B0 + 16 * D_, D_, lr, kc, half);
    acc[0][0] = WMMA_BF16(a0, b0, acc[0][0]);
    acc[0][1] = WMMA_BF16(a0, b1, acc[0][1]);
    acc[1][0] = WMMA_BF16(a1, b0, acc[1][0]);
    acc[1][1] = WMMA_BF16(a1, b1, acc[1][1]);
  }

  const float* bias = (g == 0 ? bz : (g == 1 ? br : bh)) + l * H_;
  __bf16* outp = xproj + (size_t)(g * L_ + l) * SB_ * H_;
  for (int ms = 0; ms < 2; ++ms) {
    for (int ns = 0; ns < 2; ++ns) {
      int col = nt * 32 + ns * 16 + lr;
      float bv = bias[col];
      for (int i = 0; i < 8; ++i) {
        int row = mt * 32 + ms * 16 + i + 8 * half;   // C layout: m = i + 8*half
        outp[(size_t)row * H_ + col] = (__bf16)(acc[ms][ns][i] + bv);
      }
    }
  }
}

// ---------------- grid-wide barrier ----------------
__device__ __forceinline__ void grid_barrier(unsigned* sync, unsigned nwg, unsigned& mygen) {
  __threadfence();          // make this thread's global writes device-visible
  __syncthreads();
  if (threadIdx.x == 0) {
    unsigned arrived = __hip_atomic_fetch_add(&sync[0], 1u, __ATOMIC_ACQ_REL, __HIP_MEMORY_SCOPE_AGENT);
    if (arrived == nwg - 1u) {
      __hip_atomic_store(&sync[0], 0u, __ATOMIC_RELAXED, __HIP_MEMORY_SCOPE_AGENT);
      __hip_atomic_store(&sync[1], mygen + 1u, __ATOMIC_RELEASE, __HIP_MEMORY_SCOPE_AGENT);
    } else {
      while (__hip_atomic_load(&sync[1], __ATOMIC_ACQUIRE, __HIP_MEMORY_SCOPE_AGENT) <= mygen) {
        __builtin_amdgcn_s_sleep(2);
      }
    }
  }
  mygen += 1u;
  __syncthreads();
}

// ---------------- persistent recurrent scan ----------------
// 32 WGs x 256 threads = 256 wave32s. Per step:
//   Phase A (256 waves): z/r = sigmoid(xproj + h*U^T); emit zbuf(f32), rh_bf(bf16)
//   Phase B (128 waves): hh = tanh(xproj + rh*Uh^T); h = (1-z)h + z*hh; emit output
// Each wave owns a 32(batch) x 16 tile -> every U element read from L2 once/step.
__global__ __launch_bounds__(256) void scan_kernel(
    const __bf16* __restrict__ Ubf,     // [3][L][H][H]  (z, r, h)
    const __bf16* __restrict__ xproj,   // [3][L][SB][H]
    __bf16* __restrict__ h_bf,          // [L][B][H]
    float*  __restrict__ h_f32,         // [L][B][H]
    float*  __restrict__ zbuf,          // [L][B][H]
    __bf16* __restrict__ rh_bf,         // [L][B][H]
    float*  __restrict__ out,           // [S][B][H] ++ [L][B][H]
    unsigned* sync)
{
  const int tid = threadIdx.x, wave = tid >> 5, lane = tid & 31;
  const int half = lane >> 4, lr = lane & 15;
  const int gw = blockIdx.x * 8 + wave;   // 0..255
  const unsigned NWG = 32;
  unsigned mygen = 0;

  // Phase A assignment: layer, gate(0=z,1=r), N-tile of 16
  const int lA = gw >> 7, gateA = (gw >> 6) & 1, ntA = gw & 63;
  const __bf16* UA  = Ubf + (size_t)(gateA * L_ + lA) * H_ * H_ + (size_t)(ntA * 16) * H_;
  const __bf16* hA  = h_bf + (size_t)lA * B_ * H_;
  const __bf16* xpA = xproj + (size_t)(gateA * L_ + lA) * SB_ * H_;

  // Phase B assignment (first 128 waves): layer, N-tile of 16
  const bool activeB = (gw < 128);
  const int lB = (gw >> 6) & 1, ntB = gw & 63;
  const __bf16* UB  = Ubf + (size_t)(2 * L_ + lB) * H_ * H_ + (size_t)(ntB * 16) * H_;
  const __bf16* rhB = rh_bf + (size_t)lB * B_ * H_;
  const __bf16* xpB = xproj + (size_t)(2 * L_ + lB) * SB_ * H_;

  for (int s = 0; s < S_; ++s) {
    { // ---- Phase A: z and r gates ----
      v8f acc[2] = {};
      for (int kc = 0; kc < H_; kc += 32) {
        v16bf bfrag = load_fragB(UA,           H_, lr, kc, half);
        v16bf a0    = load_fragA(hA,           H_, lr, kc, half);
        v16bf a1    = load_fragA(hA + 16 * H_, H_, lr, kc, half);
        acc[0] = WMMA_BF16(a0, bfrag, acc[0]);
        acc[1] = WMMA_BF16(a1, bfrag, acc[1]);
      }
      const int col = ntA * 16 + lr;
      for (int ms = 0; ms < 2; ++ms) {
        for (int i = 0; i < 8; ++i) {
          int b = ms * 16 + i + 8 * half;
          float pre = acc[ms][i] + (float)xpA[(size_t)(s * B_ + b) * H_ + col];
          float v = 1.0f / (1.0f + __expf(-pre));       // sigmoid
          size_t idx = (size_t)(lA * B_ + b) * H_ + col;
          if (gateA == 0) zbuf[idx] = v;                 // z gate
          else            rh_bf[idx] = (__bf16)(v * h_f32[idx]);  // r*h for Uh GEMM
        }
      }
    }
    grid_barrier(sync, NWG, mygen);

    if (activeB) { // ---- Phase B: candidate state + update ----
      v8f acc[2] = {};
      for (int kc = 0; kc < H_; kc += 32) {
        v16bf bfrag = load_fragB(UB,            H_, lr, kc, half);
        v16bf a0    = load_fragA(rhB,           H_, lr, kc, half);
        v16bf a1    = load_fragA(rhB + 16 * H_, H_, lr, kc, half);
        acc[0] = WMMA_BF16(a0, bfrag, acc[0]);
        acc[1] = WMMA_BF16(a1, bfrag, acc[1]);
      }
      const int col = ntB * 16 + lr;
      for (int ms = 0; ms < 2; ++ms) {
        for (int i = 0; i < 8; ++i) {
          int b = ms * 16 + i + 8 * half;
          float pre = acc[ms][i] + (float)xpB[(size_t)(s * B_ + b) * H_ + col];
          float hh = tanhf(pre);
          size_t idx = (size_t)(lB * B_ + b) * H_ + col;
          float hold = h_f32[idx];
          float z = zbuf[idx];
          float hnew = (1.0f - z) * hold + z * hh;
          h_f32[idx] = hnew;
          h_bf[idx] = (__bf16)hnew;
          if (lB == L_ - 1) out[(size_t)(s * B_ + b) * H_ + col] = hnew;   // output[s]
          if (s == S_ - 1)  out[(size_t)SB_ * H_ + idx] = hnew;            // h_n
        }
      }
    }
    grid_barrier(sync, NWG, mygen);
  }
}

// ---------------- host ----------------
extern "C" void kernel_launch(void* const* d_in, const int* in_sizes, int n_in,
                              void* d_out, int out_size, void* d_ws, size_t ws_size,
                              hipStream_t stream) {
  const float* x  = (const float*)d_in[0];   // [S,B,D]
  const float* h0 = (const float*)d_in[1];   // [L,B,H]
  const float* Wz = (const float*)d_in[2];   // [L,H,D]
  const float* Wr = (const float*)d_in[3];
  const float* Wh = (const float*)d_in[4];
  const float* Uz = (const float*)d_in[5];   // [L,H,H]
  const float* Ur = (const float*)d_in[6];
  const float* Uh = (const float*)d_in[7];
  const float* bz = (const float*)d_in[8];   // [L,H]
  const float* br = (const float*)d_in[9];
  const float* bh = (const float*)d_in[10];

  // workspace carve-up (256B aligned)
  char* ws = (char*)d_ws;
  size_t off = 0;
  auto take = [&](size_t bytes) -> char* {
    char* p = ws + off;
    off = (off + bytes + 255) & ~(size_t)255;
    return p;
  };
  unsigned* syncp  = (unsigned*)take(256);
  float*    h_f32  = (float*)take((size_t)L_ * B_ * H_ * 4);
  float*    zbuf   = (float*)take((size_t)L_ * B_ * H_ * 4);
  __bf16*   h_bf   = (__bf16*)take((size_t)L_ * B_ * H_ * 2);
  __bf16*   rh_bf  = (__bf16*)take((size_t)L_ * B_ * H_ * 2);
  __bf16*   x_bf   = (__bf16*)take((size_t)SB_ * D_ * 2);
  __bf16*   W_bf   = (__bf16*)take((size_t)3 * L_ * H_ * D_ * 2);
  __bf16*   U_bf   = (__bf16*)take((size_t)3 * L_ * H_ * H_ * 2);
  __bf16*   xproj  = (__bf16*)take((size_t)3 * L_ * SB_ * H_ * 2);
  (void)ws_size; (void)in_sizes; (void)n_in; (void)out_size;

  hipMemsetAsync(syncp, 0, 256, stream);

  const size_t LHD = (size_t)L_ * H_ * D_;
  const size_t LHH = (size_t)L_ * H_ * H_;
  cvt_bf16_kernel<<<4096, 256, 0, stream>>>(x,  x_bf, SB_ * D_);
  cvt_bf16_kernel<<<2048, 256, 0, stream>>>(Wz, W_bf,            (int)LHD);
  cvt_bf16_kernel<<<2048, 256, 0, stream>>>(Wr, W_bf + LHD,      (int)LHD);
  cvt_bf16_kernel<<<2048, 256, 0, stream>>>(Wh, W_bf + 2 * LHD,  (int)LHD);
  cvt_bf16_kernel<<<2048, 256, 0, stream>>>(Uz, U_bf,            (int)LHH);
  cvt_bf16_kernel<<<2048, 256, 0, stream>>>(Ur, U_bf + LHH,      (int)LHH);
  cvt_bf16_kernel<<<2048, 256, 0, stream>>>(Uh, U_bf + 2 * LHH,  (int)LHH);
  init_h_kernel<<<256, 256, 0, stream>>>(h0, h_f32, h_bf, L_ * B_ * H_);

  // 3 gates * 2 layers * 512 M-tiles * 32 N-tiles = 98304 wave-tiles / 8 waves per WG
  proj_kernel<<<12288, 256, 0, stream>>>(x_bf, W_bf, bz, br, bh, xproj);

  scan_kernel<<<32, 256, 0, stream>>>(U_bf, xproj, h_bf, h_f32, zbuf, rh_bf,
                                      (float*)d_out, syncp);
}
